// So3kratesBlock_5265629905690
// MI455X (gfx1250) — matrix operand order repaired and checked by reading, not compiled, not run on hardware
//
#include <hip/hip_runtime.h>
#include <stdint.h>

// ---------------------------------------------------------------------------
// So3krates block for MI455X (gfx1250, wave32, WMMA).
// Fused per-edge-tile MLPs with v_wmma_f32_16x16x32_bf16; f32 accumulate;
// segment-sums via L2-resident f32 atomics (node state = 10MB << 192MB L2).
// Edge-feature tiles are double-buffered into LDS with the CDNA5 async
// engine (global_load_async_to_lds_b128 / s_wait_asynccnt).
// ---------------------------------------------------------------------------

#define NN 20000
#define NE 320000
#define FDIM 128
#define NHEAD 4
#define NRBF 32
#define NLAYERS 3
#define CHID 8
#define DIN 130          // F + N_IRREPS
#define LDP 136          // LDS row stride (272B = 17*16B: keeps 16B alignment)
#define TPT 4            // edge tiles (of 16 edges) per workgroup

typedef unsigned short u16;
typedef unsigned int u32;
typedef __attribute__((ext_vector_type(16))) __bf16 v16bf;
typedef __attribute__((ext_vector_type(8)))  float  v8f;

union Frag16 { v16bf v; u16 s[16]; u32 u[8]; };
struct __align__(16) U4 { u32 a, b, c, d; };

#if __has_builtin(__builtin_amdgcn_s_wait_asynccnt)
#define AWAIT(n) __builtin_amdgcn_s_wait_asynccnt(n)
#else
#define AWAIT(n) asm volatile("s_wait_asynccnt " #n ::: "memory")
#endif

__device__ __forceinline__ u16 f2bf(float f) {
  union { float f; u32 u; } c; c.f = f;
  u32 u = c.u;
  return (u16)((u + 0x7fffu + ((u >> 16) & 1u)) >> 16);   // RNE
}
__device__ __forceinline__ float siluf(float x) {
  return x * (1.0f / (1.0f + __expf(-x)));
}
__device__ __forceinline__ v8f zero8() {
  v8f z = {0.f, 0.f, 0.f, 0.f, 0.f, 0.f, 0.f, 0.f};
  return z;
}
__device__ __forceinline__ v8f wmma_bf16(const Frag16& a, const Frag16& b, v8f c) {
  return __builtin_amdgcn_wmma_f32_16x16x32_bf16(false, a.v, false, b.v,
                                                 (short)0, c, false, false);
}
__device__ __forceinline__ void ld2x16B(Frag16& f, const u16* p0, const u16* p1) {
  U4 x = *(const U4*)p0;
  U4 y = *(const U4*)p1;
  f.u[0] = x.a; f.u[1] = x.b; f.u[2] = x.c; f.u[3] = x.d;
  f.u[4] = y.a; f.u[5] = y.b; f.u[6] = y.c; f.u[7] = y.d;
}
// A fragment (16x32 bf16) from row-major [rows x ld] global memory.
// Lane l holds row (l&15); K chunk = kc*32 + {0|8} + [0..7] and +16.
__device__ __forceinline__ Frag16 load_afrag_g(const u16* base, int row0, int ld,
                                               int kc, int lane) {
  int r = lane & 15;
  int kb = kc * 32 + ((lane >> 4) << 3);
  const u16* p = base + (size_t)(row0 + r) * ld + kb;
  Frag16 f; ld2x16B(f, p, p + 16);
  return f;
}
// A fragment from LDS tile with row stride ld (elements; ld*2 multiple of 16B).
__device__ __forceinline__ Frag16 load_afrag_lds(const u16* h, int ld, int kc,
                                                 int lane) {
  int r = lane & 15;
  int kb = kc * 32 + ((lane >> 4) << 3);
  const u16* p = h + r * ld + kb;
  Frag16 f; ld2x16B(f, p, p + 16);
  return f;
}
// B fragment from pre-packed buffer: ((kc*NT + nt)*32 + lane)*16 contiguous u16.
__device__ __forceinline__ Frag16 load_bfrag(const u16* p) {
  Frag16 f; ld2x16B(f, p, p + 8);
  return f;
}

// Async-copy one 16x32 bf16 edge-feature tile (1KB contiguous) into LDS.
// Each of the 32 lanes copies 16B; offset:512 covers the second half on both
// the LDS and global side (ISA: dsaddr = VDST+IOFFSET, maddr = VADDR+IOFFSET).
__device__ __forceinline__ void async_ef_tile(const u16* gsrc, const u16* ldst,
                                              int lane) {
  const u16* g = gsrc + lane * 8;
  u32 l = (u32)(uintptr_t)(ldst + lane * 8);
  asm volatile(
      "global_load_async_to_lds_b128 %0, %1, off\n\t"
      "global_load_async_to_lds_b128 %0, %1, off offset:512"
      :: "v"(l), "v"(g) : "memory");
}

// ---------------------------------------------------------------------------
// Small prep kernels
// ---------------------------------------------------------------------------
__global__ void init_zero_kernel(float* chi, float* dx, float* dchi) {
  int i = blockIdx.x * blockDim.x + threadIdx.x;
  if (i < NN * FDIM) dx[i] = 0.f;
  if (i < NN * CHID) { chi[i] = 0.f; dchi[i] = 0.f; }
}

__global__ void edge_prep_kernel(const float* __restrict__ ev,
                                 const float* __restrict__ dist,
                                 const float* __restrict__ cut,
                                 const int* __restrict__ recv,
                                 u16* __restrict__ ef16,
                                 float* __restrict__ sh,
                                 float* __restrict__ chi) {
  int e = blockIdx.x * blockDim.x + threadIdx.x;
  if (e >= NE) return;
  float x = ev[3 * e + 0], y = ev[3 * e + 1], z = ev[3 * e + 2];
  float inv = rsqrtf(x * x + y * y + z * z);
  x *= inv; y *= inv; z *= inv;
  const float s3 = 1.7320508075688772f;
  const float s5 = 2.2360679774997896f;
  const float s15 = 3.872983346207417f;
  float shv[8];
  shv[0] = s3 * x; shv[1] = s3 * y; shv[2] = s3 * z;
  shv[3] = s15 * x * y; shv[4] = s15 * y * z;
  shv[5] = 0.5f * s5 * (3.f * z * z - 1.f);
  shv[6] = s15 * x * z; shv[7] = 0.5f * s15 * (x * x - y * y);
  float cw = cut[e] * (1.0f / 16.0f);  // /SPHC_NORM folded
  int rv = recv[e];
#pragma unroll
  for (int c = 0; c < 8; c++) {
    sh[(size_t)e * 8 + c] = shv[c];
    atomicAdd(&chi[(size_t)rv * 8 + c], shv[c] * cw);
  }
  const float e5 = 0.006737946999085467f;          // exp(-5)
  const float dmu = (1.0f - e5) / 31.0f;
  const float bw = 2.0f * (1.0f - e5) / 32.0f;
  const float beta = 1.0f / (bw * bw);
  float tt = __expf(-dist[e]);
#pragma unroll
  for (int i = 0; i < NRBF; i++) {
    float d = tt - (e5 + dmu * i);
    ef16[(size_t)e * NRBF + i] = f2bf(__expf(-beta * d * d));
  }
}

__global__ void embed_kernel(const int* __restrict__ species,
                             const float* __restrict__ emb,
                             float* __restrict__ nf, u16* __restrict__ nf16) {
  int i = blockIdx.x * blockDim.x + threadIdx.x;
  if (i >= NN * FDIM) return;
  int n = i >> 7, c = i & 127;
  float v = emb[(size_t)species[n] * FDIM + c];
  nf[i] = v;
  nf16[i] = f2bf(v);
}

// Pack f32 [K x N] weight into WMMA-B bf16 fragment order.
__global__ void pack_b_kernel(const float* __restrict__ W, u16* __restrict__ dst,
                              int K, int N) {
  int idx = blockIdx.x * blockDim.x + threadIdx.x;
  if (idx >= K * N) return;
  int NT = N >> 4;
  int j = idx & 15;
  int lane = (idx >> 4) & 31;
  int tIdx = idx >> 9;
  int nt = tIdx % NT;
  int kc = tIdx / NT;
  int kk = kc * 32 + ((lane >> 4) << 3) + (j & 7) + ((j >> 3) << 4);   // B layout: K = {0|8}+j, +16 for j>=8
  int nn = nt * 16 + (lane & 15);
  dst[idx] = f2bf(W[(size_t)kk * N + nn]);
}

// ---------------------------------------------------------------------------
// Node GEMMs: q,k,q2,k2 = nf @ W + b.  One wave per weight matrix; 16-node tile.
// ---------------------------------------------------------------------------
struct GemmArgs {
  const u16* nf16;
  const u16* wp[4];
  const float* bias[4];
  float* out[4];
};

__global__ __launch_bounds__(128) void node_gemm_kernel(GemmArgs g) {
  int nb = blockIdx.x * 16;
  int w = threadIdx.x >> 5, lane = threadIdx.x & 31;
  int lc = lane & 15, hf = lane >> 4;
  Frag16 A[4];
#pragma unroll
  for (int kc = 0; kc < 4; kc++) A[kc] = load_afrag_g(g.nf16, nb, FDIM, kc, lane);
  const u16* wp = g.wp[w];
  const float* bias = g.bias[w];
  float* out = g.out[w];
  for (int nt = 0; nt < 8; nt++) {
    v8f acc = zero8();
#pragma unroll
    for (int kc = 0; kc < 4; kc++) {
      Frag16 B = load_bfrag(wp + ((size_t)(kc * 8 + nt) * 32 + lane) * 16);
      acc = wmma_bf16(A[kc], B, acc);
    }
    int col = nt * 16 + lc;
    float bb = bias[col];
#pragma unroll
    for (int i = 0; i < 8; i++)
      out[(size_t)(nb + i + 8 * hf) * FDIM + col] = acc[i] + bb;
  }
}

// ---------------------------------------------------------------------------
// Fused edge layer
// ---------------------------------------------------------------------------
struct EdgeArgs {
  const u16* ef16; const float* sh; const float* cut;
  const int* snd; const int* rcv;
  const float* chi; const float* nf;
  const float* q; const float* k; const float* q2; const float* k2;
  const u16* fbW1p; const float* fb_b1; const u16* fbW2p; const float* fb_b2;
  const float* fsW1; const float* fs_b1; const u16* fsW2p; const float* fs_b2;
  const u16* gbW1p; const float* gb_b1; const u16* gbW2p; const float* gb_b2;
  const float* gsW1; const float* gs_b1; const u16* gsW2p; const float* gs_b2;
  float* dx; float* dchi;
};

// w = MLP_rad(edge_feats) + MLP_sph(chi_scalar) for a 16-edge tile; result in
// accumulators: wave w owns columns [32w, 32w+32) (tiles 2w, 2w+1).
__device__ __forceinline__ void mlp_pair(
    const Frag16& Aef,
    const u16* W1p, const float* b1, const u16* W2p, const float* b2,
    const float* sW1, const float* sb1, const u16* sW2p, const float* sb2,
    const float (*csL)[2], u16* hA, u16* hB, v8f& acc0, v8f& acc1) {
  int t = threadIdx.x;
  int w = t >> 5, lane = t & 31, lc = lane & 15, hf = lane >> 4;
  // layer 1 (radial): 16x32 @ 32x128, single K-step WMMA per N-tile
#pragma unroll
  for (int s = 0; s < 2; s++) {
    int nt = 2 * w + s;
    Frag16 B = load_bfrag(W1p + ((size_t)nt * 32 + lane) * 16);
    v8f acc = zero8();
    acc = wmma_bf16(Aef, B, acc);
    int col = nt * 16 + lc;
    float bb = b1[col];
#pragma unroll
    for (int i = 0; i < 8; i++)
      hA[(i + 8 * hf) * LDP + col] = f2bf(siluf(acc[i] + bb));
  }
  // layer 1 (sphc scalars): 2 -> 128 on VALU
  {
    int col = t;
    float w0 = sW1[col], w1 = sW1[FDIM + col], bb = sb1[col];
#pragma unroll
    for (int r = 0; r < 16; r++) {
      float v = csL[r][0] * w0 + csL[r][1] * w1 + bb;
      hB[r * LDP + col] = f2bf(siluf(v));
    }
  }
  __syncthreads();
  // layer 2: (16x128 @ 128x128)_rad + (16x128 @ 128x128)_sph, shared accumulator
#pragma unroll
  for (int s = 0; s < 2; s++) {
    int nt = 2 * w + s;
    v8f acc = zero8();
#pragma unroll
    for (int kc = 0; kc < 4; kc++) {
      Frag16 A1 = load_afrag_lds(hA, LDP, kc, lane);
      Frag16 B1 = load_bfrag(W2p + ((size_t)(kc * 8 + nt) * 32 + lane) * 16);
      acc = wmma_bf16(A1, B1, acc);
      Frag16 A2 = load_afrag_lds(hB, LDP, kc, lane);
      Frag16 B2 = load_bfrag(sW2p + ((size_t)(kc * 8 + nt) * 32 + lane) * 16);
      acc = wmma_bf16(A2, B2, acc);
    }
    int col = nt * 16 + lc;
    float bb = b2[col] + sb2[col];
#pragma unroll
    for (int i = 0; i < 8; i++) acc[i] += bb;
    if (s == 0) acc0 = acc; else acc1 = acc;
  }
  __syncthreads();
}

__global__ __launch_bounds__(128) void edge_layer_kernel(EdgeArgs a) {
  __shared__ u16 efS[2][16 * NRBF];   // 2KB async double buffer
  __shared__ u16 hA[16 * LDP];
  __shared__ u16 hB[16 * LDP];
  __shared__ int sndL[16], rcvL[16];
  __shared__ float cutL[16];
  __shared__ float shL[16][8];
  __shared__ float csL[16][2];
  __shared__ float alphaL[16][NHEAD];
  __shared__ float betP[4][16];
  __shared__ float betL[16][2];

  const int t = threadIdx.x;
  const int w = t >> 5, lane = t & 31, lc = lane & 15, hf = lane >> 4;
  const int eBase = blockIdx.x * (16 * TPT);

  // kick off async staging of tile 0's radial features (wave 0 issues)
  if (w == 0)
    async_ef_tile(a.ef16 + (size_t)eBase * NRBF, &efS[0][0], lane);

  for (int tt = 0; tt < TPT; ++tt) {
    const int e0 = eBase + tt * 16;
    const u16* efL = &efS[tt & 1][0];

    __syncthreads();   // previous iteration's LDS consumers are done
    if (w == 0 && tt + 1 < TPT)
      async_ef_tile(a.ef16 + (size_t)(e0 + 16) * NRBF, &efS[(tt + 1) & 1][0], lane);

    if (t < 16) {
      int e = e0 + t;
      sndL[t] = a.snd[e]; rcvL[t] = a.rcv[e]; cutL[t] = a.cut[e];
    }
    { int r = t >> 3, c = t & 7; shL[r][c] = a.sh[(size_t)(e0 + r) * 8 + c]; }
    __syncthreads();
    if (t < 16) {
      const float* cs = a.chi + (size_t)sndL[t] * CHID;
      const float* cr = a.chi + (size_t)rcvL[t] * CHID;
      float s0 = 0.f, s1 = 0.f;
#pragma unroll
      for (int c = 0; c < CHID; c++) {
        float d = cs[c] - cr[c];
        if (c < 3) s0 += d * d; else s1 += d * d;
      }
      csL[t][0] = s0; csL[t][1] = s1;
    }
    // loads complete in order: with the next tile in flight (2 ops), waiting
    // for <=2 outstanding guarantees the current tile (issued earlier) landed.
    if (w == 0) {
      if (tt + 1 < TPT) { AWAIT(2); } else { AWAIT(0); }
    }
    __syncthreads();

    Frag16 Aef = load_afrag_lds(efL, NRBF, 0, lane);
    v8f acc0, acc1;

    // ---- feature attention block ----
    mlp_pair(Aef, a.fbW1p, a.fb_b1, a.fbW2p, a.fb_b2,
             a.fsW1, a.fs_b1, a.fsW2p, a.fs_b2, csL, hA, hB, acc0, acc1);
    {
      const int col0 = 32 * w + lc, col1 = col0 + 16;  // wave w == head w
      const float invs = 0.17677669529663687f;          // 1/sqrt(dh=32)
#pragma unroll
      for (int i = 0; i < 8; i++) {
        int r = i + 8 * hf;
        int rs = sndL[r], rr = rcvL[r];
        float pa = a.q[(size_t)rr * FDIM + col0] * a.k[(size_t)rs * FDIM + col0] * acc0[i]
                 + a.q[(size_t)rr * FDIM + col1] * a.k[(size_t)rs * FDIM + col1] * acc1[i];
        pa += __shfl_xor(pa, 1, 32);
        pa += __shfl_xor(pa, 2, 32);
        pa += __shfl_xor(pa, 4, 32);
        pa += __shfl_xor(pa, 8, 32);
        if (lc == 0) alphaL[r][w] = pa * invs * cutL[r] * (1.0f / 16.0f);
      }
    }
    __syncthreads();
    {
      int col = t, head = t >> 5;
#pragma unroll 4
      for (int r = 0; r < 16; r++) {
        float av = alphaL[r][head];
        atomicAdd(&a.dx[(size_t)rcvL[r] * FDIM + col],
                  av * a.nf[(size_t)sndL[r] * FDIM + col]);
      }
    }
    __syncthreads();

    // ---- geometric attention block ----
    mlp_pair(Aef, a.gbW1p, a.gb_b1, a.gbW2p, a.gb_b2,
             a.gsW1, a.gs_b1, a.gsW2p, a.gs_b2, csL, hA, hB, acc0, acc1);
    {
      const int col0 = 32 * w + lc, col1 = col0 + 16;
#pragma unroll
      for (int i = 0; i < 8; i++) {
        int r = i + 8 * hf;
        int rs = sndL[r], rr = rcvL[r];
        float pa = a.q2[(size_t)rr * FDIM + col0] * a.k2[(size_t)rs * FDIM + col0] * acc0[i]
                 + a.q2[(size_t)rr * FDIM + col1] * a.k2[(size_t)rs * FDIM + col1] * acc1[i];
        pa += __shfl_xor(pa, 1, 32);
        pa += __shfl_xor(pa, 2, 32);
        pa += __shfl_xor(pa, 4, 32);
        pa += __shfl_xor(pa, 8, 32);
        if (lc == 0) betP[w][r] = pa;
      }
    }
    __syncthreads();
    if (t < 32) {
      int r = t & 15, ir = t >> 4;   // irreps span wave pairs (0,1) and (2,3)
      betL[r][ir] = (betP[2 * ir][r] + betP[2 * ir + 1][r]) * 0.125f  // 1/sqrt(64)
                    * cutL[r] * (1.0f / 16.0f);
    }
    __syncthreads();
    {
      int r = t >> 3, c = t & 7;
      float bf = betL[r][(c < 3) ? 0 : 1];   // REP = [3,5]
      atomicAdd(&a.dchi[(size_t)rcvL[r] * CHID + c], bf * shL[r][c]);
    }
  }
}

// ---------------------------------------------------------------------------
// Node update: nf += dx; chi += dchi; y = silu([nf, |chi|^2] @ Wint + b);
// nf += y[:F]; chi *= (1 + rep(y[F:])). Also re-zeros dx/dchi for next layer.
// ---------------------------------------------------------------------------
struct UpdArgs {
  float* nf; u16* nf16; float* chi; float* dx; float* dchi;
  const float* Wint; const float* bint;
};

__global__ __launch_bounds__(128) void node_update_kernel(UpdArgs a) {
  __shared__ float xs[DIN];
  __shared__ float ys[DIN];
  __shared__ float chiS[CHID];
  int n = blockIdx.x, t = threadIdx.x;
  size_t nfo = (size_t)n * FDIM + t;
  float nf_new = a.nf[nfo] + a.dx[nfo];
  xs[t] = nf_new;
  a.dx[nfo] = 0.f;
  if (t < CHID) {
    size_t co = (size_t)n * CHID + t;
    float cv = a.chi[co] + a.dchi[co];
    chiS[t] = cv;
    a.dchi[co] = 0.f;
  }
  __syncthreads();
  if (t < 2) {
    float s = 0.f;
    int b = t ? 3 : 0, e = t ? 8 : 3;
    for (int c = b; c < e; c++) s += chiS[c] * chiS[c];
    xs[FDIM + t] = s;
  }
  __syncthreads();
  for (int o = t; o < DIN; o += 128) {
    float acc = a.bint[o];
    for (int j = 0; j < DIN; j++) acc += xs[j] * a.Wint[(size_t)j * DIN + o];
    ys[o] = siluf(acc);
  }
  __syncthreads();
  float nf2 = nf_new + ys[t];
  a.nf[nfo] = nf2;
  a.nf16[nfo] = f2bf(nf2);
  if (t < CHID) {
    int ir = (t < 3) ? 0 : 1;
    a.chi[(size_t)n * CHID + t] = chiS[t] * (1.0f + ys[FDIM + ir]);
  }
}

// ---------------------------------------------------------------------------
// Energy MLP: [F,F,F,1] with silu between layers; WMMA for both 128x128 GEMMs.
// ---------------------------------------------------------------------------
struct EnergyArgs {
  const u16* nf16; const u16* E0p; const float* b0;
  const u16* E1p; const float* b1;
  const float* E2; const float* b2; float* out;
};

__global__ __launch_bounds__(128) void energy_kernel(EnergyArgs a) {
  __shared__ u16 hA[16 * LDP];
  __shared__ float h2[16][LDP];
  __shared__ float parts[16][8];
  int nb = blockIdx.x * 16;
  int t = threadIdx.x, w = t >> 5, lane = t & 31, lc = lane & 15, hf = lane >> 4;
  Frag16 A[4];
#pragma unroll
  for (int kc = 0; kc < 4; kc++) A[kc] = load_afrag_g(a.nf16, nb, FDIM, kc, lane);
#pragma unroll
  for (int s = 0; s < 2; s++) {
    int nt = 2 * w + s;
    v8f acc = zero8();
#pragma unroll
    for (int kc = 0; kc < 4; kc++) {
      Frag16 B = load_bfrag(a.E0p + ((size_t)(kc * 8 + nt) * 32 + lane) * 16);
      acc = wmma_bf16(A[kc], B, acc);
    }
    int col = nt * 16 + lc;
    float bb = a.b0[col];
#pragma unroll
    for (int i = 0; i < 8; i++)
      hA[(i + 8 * hf) * LDP + col] = f2bf(siluf(acc[i] + bb));
  }
  __syncthreads();
#pragma unroll
  for (int s = 0; s < 2; s++) {
    int nt = 2 * w + s;
    v8f acc = zero8();
#pragma unroll
    for (int kc = 0; kc < 4; kc++) {
      Frag16 AL = load_afrag_lds(hA, LDP, kc, lane);
      Frag16 B = load_bfrag(a.E1p + ((size_t)(kc * 8 + nt) * 32 + lane) * 16);
      acc = wmma_bf16(AL, B, acc);
    }
    int col = nt * 16 + lc;
    float bb = a.b1[col];
#pragma unroll
    for (int i = 0; i < 8; i++)
      h2[i + 8 * hf][col] = siluf(acc[i] + bb);
  }
  __syncthreads();
  {
    int r = t & 15, ch = t >> 4;
    float p = 0.f;
#pragma unroll
    for (int j = 0; j < 16; j++) p += h2[r][ch * 16 + j] * a.E2[ch * 16 + j];
    parts[r][ch] = p;
  }
  __syncthreads();
  if (t < 16) {
    float e = a.b2[0];
#pragma unroll
    for (int c = 0; c < 8; c++) e += parts[t][c];
    a.out[nb + t] = e;
  }
}

// ---------------------------------------------------------------------------
// Host orchestration
// Param pytree flatten order assumed: top-level in setup_inputs() insertion
// order; nested params dict in sorted-key order:
//   6: embed, 7..12: energy (W,b)x3, layer l base B=13+26l:
//   Wint(W,b) Wk Wk2 Wq Wq2 fb_rad(W1,b1,W2,b2) fb_sph gb_rad gb_sph
// ---------------------------------------------------------------------------
extern "C" void kernel_launch(void* const* d_in, const int* in_sizes, int n_in,
                              void* d_out, int out_size, void* d_ws, size_t ws_size,
                              hipStream_t stream) {
  (void)in_sizes; (void)n_in; (void)out_size; (void)ws_size;
  auto F32 = [&](int i) { return (const float*)d_in[i]; };
  auto I32 = [&](int i) { return (const int*)d_in[i]; };

  const float* edge_vectors = F32(0);
  const float* distances = F32(1);
  const float* cutoffs = F32(2);
  const int* node_species = I32(3);
  const int* senders = I32(4);
  const int* receivers = I32(5);
  const float* embed = F32(6);

  size_t off = 0;
  auto alloc = [&](size_t bytes) -> void* {
    void* p = (void*)((char*)d_ws + off);
    off += (bytes + 255) & ~(size_t)255;
    return p;
  };
  u16* ef16 = (u16*)alloc((size_t)NE * NRBF * 2);
  float* sh = (float*)alloc((size_t)NE * CHID * 4);
  float* nf = (float*)alloc((size_t)NN * FDIM * 4);
  u16* nf16 = (u16*)alloc((size_t)NN * FDIM * 2);
  float* chi = (float*)alloc((size_t)NN * CHID * 4);
  float* qb = (float*)alloc((size_t)NN * FDIM * 4);
  float* kb = (float*)alloc((size_t)NN * FDIM * 4);
  float* q2b = (float*)alloc((size_t)NN * FDIM * 4);
  float* k2b = (float*)alloc((size_t)NN * FDIM * 4);
  float* dx = (float*)alloc((size_t)NN * FDIM * 4);
  float* dchi = (float*)alloc((size_t)NN * CHID * 4);

  u16 *fbW1p[NLAYERS], *fbW2p[NLAYERS], *fsW2p[NLAYERS];
  u16 *gbW1p[NLAYERS], *gbW2p[NLAYERS], *gsW2p[NLAYERS];
  u16 *Wqp[NLAYERS], *Wkp[NLAYERS], *Wq2p[NLAYERS], *Wk2p[NLAYERS];
  for (int l = 0; l < NLAYERS; l++) {
    fbW1p[l] = (u16*)alloc((size_t)32 * FDIM * 2);
    fbW2p[l] = (u16*)alloc((size_t)FDIM * FDIM * 2);
    fsW2p[l] = (u16*)alloc((size_t)FDIM * FDIM * 2);
    gbW1p[l] = (u16*)alloc((size_t)32 * FDIM * 2);
    gbW2p[l] = (u16*)alloc((size_t)FDIM * FDIM * 2);
    gsW2p[l] = (u16*)alloc((size_t)FDIM * FDIM * 2);
    Wqp[l] = (u16*)alloc((size_t)FDIM * FDIM * 2);
    Wkp[l] = (u16*)alloc((size_t)FDIM * FDIM * 2);
    Wq2p[l] = (u16*)alloc((size_t)FDIM * FDIM * 2);
    Wk2p[l] = (u16*)alloc((size_t)FDIM * FDIM * 2);
  }
  u16* E0p = (u16*)alloc((size_t)FDIM * FDIM * 2);
  u16* E1p = (u16*)alloc((size_t)FDIM * FDIM * 2);

  auto pack = [&](const float* W, u16* dst, int K, int N) {
    int total = K * N;
    pack_b_kernel<<<(total + 255) / 256, 256, 0, stream>>>(W, dst, K, N);
  };

  init_zero_kernel<<<(NN * FDIM + 255) / 256, 256, 0, stream>>>(chi, dx, dchi);
  edge_prep_kernel<<<(NE + 255) / 256, 256, 0, stream>>>(
      edge_vectors, distances, cutoffs, receivers, ef16, sh, chi);
  embed_kernel<<<(NN * FDIM + 255) / 256, 256, 0, stream>>>(node_species, embed, nf, nf16);

  for (int l = 0; l < NLAYERS; l++) {
    int B = 13 + 26 * l;
    pack(F32(B + 10), fbW1p[l], 32, FDIM);     // fb_rad W1
    pack(F32(B + 12), fbW2p[l], FDIM, FDIM);   // fb_rad W2
    pack(F32(B + 16), fsW2p[l], FDIM, FDIM);   // fb_sph W2
    pack(F32(B + 18), gbW1p[l], 32, FDIM);     // gb_rad W1
    pack(F32(B + 20), gbW2p[l], FDIM, FDIM);   // gb_rad W2
    pack(F32(B + 24), gsW2p[l], FDIM, FDIM);   // gb_sph W2
    pack(F32(B + 6), Wqp[l], FDIM, FDIM);      // Wq
    pack(F32(B + 2), Wkp[l], FDIM, FDIM);      // Wk
    pack(F32(B + 8), Wq2p[l], FDIM, FDIM);     // Wq2
    pack(F32(B + 4), Wk2p[l], FDIM, FDIM);     // Wk2
  }
  pack(F32(7), E0p, FDIM, FDIM);
  pack(F32(9), E1p, FDIM, FDIM);

  for (int l = 0; l < NLAYERS; l++) {
    int B = 13 + 26 * l;
    GemmArgs ga;
    ga.nf16 = nf16;
    ga.wp[0] = Wqp[l];  ga.wp[1] = Wkp[l];  ga.wp[2] = Wq2p[l]; ga.wp[3] = Wk2p[l];
    ga.bias[0] = F32(B + 7); ga.bias[1] = F32(B + 3);
    ga.bias[2] = F32(B + 9); ga.bias[3] = F32(B + 5);
    ga.out[0] = qb; ga.out[1] = kb; ga.out[2] = q2b; ga.out[3] = k2b;
    node_gemm_kernel<<<NN / 16, 128, 0, stream>>>(ga);

    EdgeArgs ea;
    ea.ef16 = ef16; ea.sh = sh; ea.cut = cutoffs;
    ea.snd = senders; ea.rcv = receivers;
    ea.chi = chi; ea.nf = nf;
    ea.q = qb; ea.k = kb; ea.q2 = q2b; ea.k2 = k2b;
    ea.fbW1p = fbW1p[l]; ea.fb_b1 = F32(B + 11); ea.fbW2p = fbW2p[l]; ea.fb_b2 = F32(B + 13);
    ea.fsW1 = F32(B + 14); ea.fs_b1 = F32(B + 15); ea.fsW2p = fsW2p[l]; ea.fs_b2 = F32(B + 17);
    ea.gbW1p = gbW1p[l]; ea.gb_b1 = F32(B + 19); ea.gbW2p = gbW2p[l]; ea.gb_b2 = F32(B + 21);
    ea.gsW1 = F32(B + 22); ea.gs_b1 = F32(B + 23); ea.gsW2p = gsW2p[l]; ea.gs_b2 = F32(B + 25);
    ea.dx = dx; ea.dchi = dchi;
    edge_layer_kernel<<<NE / (16 * TPT), 128, 0, stream>>>(ea);

    UpdArgs ua;
    ua.nf = nf; ua.nf16 = nf16; ua.chi = chi; ua.dx = dx; ua.dchi = dchi;
    ua.Wint = F32(B + 0); ua.bint = F32(B + 1);
    node_update_kernel<<<NN, 128, 0, stream>>>(ua);
  }

  EnergyArgs na;
  na.nf16 = nf16; na.E0p = E0p; na.b0 = F32(8);
  na.E1p = E1p; na.b1 = F32(10);
  na.E2 = F32(11); na.b2 = F32(12);
  na.out = (float*)d_out;
  energy_kernel<<<NN / 16, 128, 0, stream>>>(na);
}
